// TransformerBlock_1005022347979
// MI455X (gfx1250) — matrix operand order repaired and checked by reading, not compiled, not run on hardware
//
#include <hip/hip_runtime.h>
#include <hip/hip_bf16.h>

typedef __bf16 bf16_t;
typedef __attribute__((ext_vector_type(16))) __bf16 v16bf;
typedef __attribute__((ext_vector_type(8)))  __bf16 v8bf;
typedef __attribute__((ext_vector_type(8)))  float  v8f;
typedef int v4i_vs __attribute__((vector_size(16)));   // matches builtin prototype

#define TSEQ  2048
#define NBATCH 2
#define NH    16
#define HD    64
#define CDIM  1024
#define MROWS 4096   // B*T

// Async global->LDS path (CDNA5 GLOBAL_LOAD_ASYNC_TO_LDS_B128, ASYNCcnt).
#if defined(__HIP_DEVICE_COMPILE__) && __has_builtin(__builtin_amdgcn_global_load_async_to_lds_b128)
#define HAVE_ASYNC_LDS 1
#else
#define HAVE_ASYNC_LDS 0
#endif

__device__ __forceinline__ void async_copy16(const bf16_t* g, bf16_t* l) {
#if HAVE_ASYNC_LDS
  __builtin_amdgcn_global_load_async_to_lds_b128(
      (__attribute__((address_space(1))) v4i_vs*)(g),
      (__attribute__((address_space(3))) v4i_vs*)(l), 0, 0);
#else
  *(uint4*)l = *(const uint4*)g;
#endif
}

__device__ __forceinline__ void wait_async_all() {
#if HAVE_ASYNC_LDS
#if __has_builtin(__builtin_amdgcn_s_wait_asynccnt)
  __builtin_amdgcn_s_wait_asynccnt(0);
#else
  asm volatile("s_wait_asynccnt 0x0" ::: "memory");
#endif
#endif
}

// ---------------------------------------------------------------------------
// WMMA fragment loaders (wave32, v_wmma_f32_16x16x32_bf16 layouts, ISA 7.12.2)
// A (16x32, MxK): lane L holds M=L%16; K = kb..kb+7 (v0-3) and kb+16..kb+23
//                 (v4-7), kb = (L>=16 ? 8 : 0).
// B (32x16, KxN): lane L holds N=L%16; K = kb..kb+7 and kb+16..kb+23, kb = (L>=16?8:0).
// ---------------------------------------------------------------------------
__device__ __forceinline__ v16bf load_a_frag(const bf16_t* rowp, int lane) {
  const int kb = (lane & 16) ? 8 : 0;
  union { v16bf v; v8bf h[2]; } u;
  u.h[0] = *(const v8bf*)(rowp + kb);
  u.h[1] = *(const v8bf*)(rowp + kb + 16);
  return u.v;
}
__device__ __forceinline__ v16bf load_b_frag(const bf16_t* colp, int lane) {
  const int ks = (lane & 16) ? 8 : 0;
  union { v16bf v; v8bf h[2]; } u;
  u.h[0] = *(const v8bf*)(colp + ks);
  u.h[1] = *(const v8bf*)(colp + ks + 16);
  return u.v;
}
__device__ __forceinline__ v8f wmma_bf16(v16bf a, v16bf b, v8f c) {
  return __builtin_amdgcn_wmma_f32_16x16x32_bf16(false, a, false, b,
                                                 (short)0, c, false, false);
}

// ---------------------------------------------------------------------------
// Weight transpose + fp32 -> bf16 convert:  W[K][N] -> Wt[N][K]
// ---------------------------------------------------------------------------
__global__ void __launch_bounds__(256)
wt_kernel(const float* __restrict__ W, bf16_t* __restrict__ Wt, int K, int N) {
  __shared__ float tile[32][33];
  const int tx = threadIdx.x, ty = threadIdx.y;
  const int nb = blockIdx.x * 32, kb = blockIdx.y * 32;
#pragma unroll
  for (int i = 0; i < 4; ++i)
    tile[ty + i * 8][tx] = W[(size_t)(kb + ty + i * 8) * N + nb + tx];
  __syncthreads();
#pragma unroll
  for (int i = 0; i < 4; ++i)
    Wt[(size_t)(nb + ty + i * 8) * K + kb + tx] = (bf16_t)tile[tx][ty + i * 8];
}

// ---------------------------------------------------------------------------
// LayerNorm (fp32 in -> bf16 out), one block per row of 1024
// ---------------------------------------------------------------------------
__global__ void __launch_bounds__(256)
ln_kernel(const float* __restrict__ X, const float* __restrict__ g,
          const float* __restrict__ bta, bf16_t* __restrict__ out) {
  __shared__ float rs[256], rs2[256];
  const int row = blockIdx.x, tid = threadIdx.x;
  const float* x = X + (size_t)row * CDIM;
  float vals[4], s = 0.f, s2 = 0.f;
#pragma unroll
  for (int i = 0; i < 4; ++i) {
    float v = x[tid + i * 256];
    vals[i] = v; s += v; s2 += v * v;
  }
  rs[tid] = s; rs2[tid] = s2; __syncthreads();
  for (int off = 128; off > 0; off >>= 1) {
    if (tid < off) { rs[tid] += rs[tid + off]; rs2[tid] += rs2[tid + off]; }
    __syncthreads();
  }
  const float mean = rs[0] * (1.f / CDIM);
  const float var  = rs2[0] * (1.f / CDIM) - mean * mean;
  const float rstd = rsqrtf(var + 1e-5f);
#pragma unroll
  for (int i = 0; i < 4; ++i) {
    const int c = tid + i * 256;
    out[(size_t)row * CDIM + c] = (bf16_t)((vals[i] - mean) * rstd * g[c] + bta[c]);
  }
}

// ---------------------------------------------------------------------------
// bf16 WMMA GEMM: C[M,N] = A[M,K] * Wt[N,K]^T + bias, with epilogues
// Tile: 128x128x32, 256 threads (8 waves, 4Mx2N), each wave: 2x4 16x16 tiles.
// Double-buffered LDS; next K-tile staged with async global->LDS while the
// current tile feeds the WMMAs (one barrier + one s_wait_asynccnt per step).
// EPI: 0 = bf16 out, 1 = relu->bf16 out, 2 = fp32 out + residual,
//      3 = scatter into Q[b,h,t,d], K[b,h,t,d], Vt[b,h,d,t]
// ---------------------------------------------------------------------------
#define BM 128
#define BN 128
#define BK 32
#define LDA 40  // bf16 elems per LDS row (32 + 8 pad -> 80B: 16B-aligned, conflict-free)

template <int EPI>
__global__ void __launch_bounds__(256)
gemm_bf16_kernel(const bf16_t* __restrict__ A, const bf16_t* __restrict__ Bw,
                 const float* __restrict__ bias, int N, int K,
                 bf16_t* __restrict__ outB, float* __restrict__ outF,
                 const float* __restrict__ resid,
                 bf16_t* __restrict__ Qo, bf16_t* __restrict__ Ko,
                 bf16_t* __restrict__ Vto) {
  __shared__ bf16_t sA[2][BM * LDA];
  __shared__ bf16_t sB[2][BN * LDA];
  const int tid = threadIdx.x;
  const int lane = tid & 31, wave = tid >> 5;
  const int wm = wave & 3, wn = wave >> 2;
  const int lm = lane & 15;
  const int m0 = blockIdx.y * BM;
  const int n0 = blockIdx.x * BN;

  v8f acc[2][4];
#pragma unroll
  for (int i = 0; i < 2; ++i)
#pragma unroll
    for (int j = 0; j < 4; ++j)
#pragma unroll
      for (int r = 0; r < 8; ++r) acc[i][j][r] = 0.f;

  // stage one 128x32 A tile + 128x32 B tile into LDS buffer `buf`
  auto stage = [&](int buf, int kt) {
#pragma unroll
    for (int c = 0; c < 2; ++c) {
      const int chunk = tid * 2 + c;   // 0..511
      const int row = chunk >> 2;
      const int c8 = chunk & 3;
      async_copy16(A + (size_t)(m0 + row) * K + kt * BK + c8 * 8,
                   &sA[buf][row * LDA + c8 * 8]);
      async_copy16(Bw + (size_t)(n0 + row) * K + kt * BK + c8 * 8,
                   &sB[buf][row * LDA + c8 * 8]);
    }
  };

  const int nk = K / BK;
  stage(0, 0);
  for (int kt = 0; kt < nk; ++kt) {
    const int cur = kt & 1;
    wait_async_all();      // my async writes for `cur` have landed
    __syncthreads();       // everyone's have; prev buffer no longer read
    if (kt + 1 < nk) stage(cur ^ 1, kt + 1);  // overlap with compute below

    v16bf af[2], bfr[4];
#pragma unroll
    for (int i = 0; i < 2; ++i)
      af[i] = load_a_frag(&sA[cur][(wm * 32 + i * 16 + lm) * LDA], lane);
#pragma unroll
    for (int j = 0; j < 4; ++j)
      bfr[j] = load_b_frag(&sB[cur][(wn * 64 + j * 16 + lm) * LDA], lane);
#pragma unroll
    for (int i = 0; i < 2; ++i)
#pragma unroll
      for (int j = 0; j < 4; ++j)
        acc[i][j] = wmma_bf16(af[i], bfr[j], acc[i][j]);
  }

  // Epilogue: C tile layout — lane holds col n = lm; VGPR r -> row r (lanes
  // 0-15) / r+8 (lanes 16-31)
  const int rofs = (lane & 16) ? 8 : 0;
#pragma unroll
  for (int i = 0; i < 2; ++i) {
#pragma unroll
    for (int j = 0; j < 4; ++j) {
      const int col = n0 + wn * 64 + j * 16 + lm;
      const float bcol = bias[col];
#pragma unroll
      for (int r = 0; r < 8; ++r) {
        const int m = m0 + wm * 32 + i * 16 + r + rofs;
        const float v = acc[i][j][r] + bcol;
        if (EPI == 0) {
          outB[(size_t)m * N + col] = (bf16_t)v;
        } else if (EPI == 1) {
          outB[(size_t)m * N + col] = (bf16_t)fmaxf(v, 0.f);
        } else if (EPI == 2) {
          outF[(size_t)m * N + col] = resid[(size_t)m * N + col] + v;
        } else {  // EPI == 3: QKV scatter
          const int t = m & (TSEQ - 1);
          const int b = m >> 11;
          if (col < CDIM) {
            const int hh = col >> 6, d = col & 63;
            Qo[(((size_t)(b * NH + hh)) * TSEQ + t) * HD + d] = (bf16_t)v;
          } else if (col < 2 * CDIM) {
            const int c2 = col - CDIM, hh = c2 >> 6, d = c2 & 63;
            Ko[(((size_t)(b * NH + hh)) * TSEQ + t) * HD + d] = (bf16_t)v;
          } else {
            const int c2 = col - 2 * CDIM, hh = c2 >> 6, d = c2 & 63;
            Vto[(((size_t)(b * NH + hh)) * HD + d) * TSEQ + t] = (bf16_t)v;
          }
        }
      }
    }
  }
}

// ---------------------------------------------------------------------------
// Causal flash attention. Grid: (T/64, B*H). 128 threads = 4 waves; each wave
// owns 16 query rows. Q[b,h,t,d], K[b,h,t,d], Vt[b,h,d,t] (all bf16).
// Output Y[b*T+t][h*64+d] bf16 (already merged-head layout for proj GEMM).
// ---------------------------------------------------------------------------
__global__ void __launch_bounds__(128)
attn_kernel(const bf16_t* __restrict__ Qm, const bf16_t* __restrict__ Km,
            const bf16_t* __restrict__ Vtm, bf16_t* __restrict__ Y) {
  __shared__ bf16_t sQ[64 * 72];
  __shared__ bf16_t sK[64 * 72];
  __shared__ bf16_t sV[64 * 72];   // sV[d][key]
  __shared__ float  sS[64 * 68];
  __shared__ bf16_t sP[64 * 72];
  __shared__ float  sAlpha[64];

  const int tid = threadIdx.x, lane = tid & 31, wave = tid >> 5;
  const int qt = blockIdx.x, bh = blockIdx.y;
  const int b = bh >> 4, h = bh & 15;
  const int q0 = qt * 64;
  const int lm = lane & 15;
  const int qr = wave * 16;
  const int rofs = (lane & 16) ? 8 : 0;

  // stage the 64x64 Q tile once
  const bf16_t* Qbase = Qm + ((size_t)bh * TSEQ + q0) * HD;
#pragma unroll
  for (int c = 0; c < 4; ++c) {
    const int chunk = tid * 4 + c;       // 0..511
    const int row = chunk >> 3, c8 = chunk & 7;
    *(uint4*)(&sQ[row * 72 + c8 * 8]) =
        *(const uint4*)(Qbase + (size_t)row * HD + c8 * 8);
  }

  float rowm = -INFINITY, rowl = 0.f;    // live in threads 0..63
  v8f accO[4];
#pragma unroll
  for (int j = 0; j < 4; ++j)
#pragma unroll
    for (int r = 0; r < 8; ++r) accO[j][r] = 0.f;

  for (int kb = 0; kb <= qt; ++kb) {
    const int k0 = kb * 64;
    __syncthreads();  // protect LDS reuse (and Q staging on first iter)
    const bf16_t* Kbase = Km + ((size_t)bh * TSEQ + k0) * HD;
    const bf16_t* Vbase = Vtm + (size_t)bh * HD * TSEQ + k0;
#pragma unroll
    for (int c = 0; c < 4; ++c) {
      const int chunk = tid * 4 + c;
      const int row = chunk >> 3, c8 = chunk & 7;
      *(uint4*)(&sK[row * 72 + c8 * 8]) =
          *(const uint4*)(Kbase + (size_t)row * HD + c8 * 8);
      *(uint4*)(&sV[row * 72 + c8 * 8]) =
          *(const uint4*)(Vbase + (size_t)row * TSEQ + c8 * 8);
    }
    __syncthreads();

    // S = (Q K^T) * hd^-0.5, causal-masked
    v8f accS[4];
#pragma unroll
    for (int j = 0; j < 4; ++j)
#pragma unroll
      for (int r = 0; r < 8; ++r) accS[j][r] = 0.f;
#pragma unroll
    for (int ks = 0; ks < 2; ++ks) {
      const v16bf aq = load_a_frag(&sQ[(qr + lm) * 72 + ks * 32], lane);
#pragma unroll
      for (int j = 0; j < 4; ++j) {
        const v16bf bk = load_b_frag(&sK[(j * 16 + lm) * 72 + ks * 32], lane);
        accS[j] = wmma_bf16(aq, bk, accS[j]);
      }
    }
#pragma unroll
    for (int j = 0; j < 4; ++j) {
      const int n = j * 16 + lm;
#pragma unroll
      for (int r = 0; r < 8; ++r) {
        const int ql = qr + r + rofs;
        float v = accS[j][r] * 0.125f;           // hd^-0.5, hd = 64
        if (k0 + n > q0 + ql) v = -INFINITY;     // causal
        sS[ql * 68 + n] = v;
      }
    }
    __syncthreads();

    // online softmax (one thread per query row)
    if (tid < 64) {
      float mx = -INFINITY;
      for (int k = 0; k < 64; ++k) mx = fmaxf(mx, sS[tid * 68 + k]);
      const float mnew = fmaxf(rowm, mx);
      const float alpha = __expf(rowm - mnew);
      float sum = 0.f;
      for (int k = 0; k < 64; ++k) {
        const float p = __expf(sS[tid * 68 + k] - mnew);
        sP[tid * 72 + k] = (bf16_t)p;
        sum += p;
      }
      rowl = rowl * alpha + sum;
      rowm = mnew;
      sAlpha[tid] = alpha;
    }
    __syncthreads();

    // rescale O, then O += P * V
#pragma unroll
    for (int j = 0; j < 4; ++j)
#pragma unroll
      for (int r = 0; r < 8; ++r) accO[j][r] *= sAlpha[qr + r + rofs];
#pragma unroll
    for (int ks = 0; ks < 2; ++ks) {
      const v16bf ap = load_a_frag(&sP[(qr + lm) * 72 + ks * 32], lane);
#pragma unroll
      for (int j = 0; j < 4; ++j) {
        const v16bf bv = load_b_frag(&sV[(j * 16 + lm) * 72 + ks * 32], lane);
        accO[j] = wmma_bf16(ap, bv, accO[j]);
      }
    }
  }

  __syncthreads();
  if (tid < 64) sAlpha[tid] = 1.0f / rowl;
  __syncthreads();
#pragma unroll
  for (int j = 0; j < 4; ++j) {
    const int d = j * 16 + lm;
#pragma unroll
    for (int r = 0; r < 8; ++r) {
      const int ql = qr + r + rofs;
      const float v = accO[j][r] * sAlpha[ql];
      Y[((size_t)(b * TSEQ + q0 + ql)) * CDIM + h * HD + d] = (bf16_t)v;
    }
  }
}

// ---------------------------------------------------------------------------
// Host launcher
// ---------------------------------------------------------------------------
extern "C" void kernel_launch(void* const* d_in, const int* in_sizes, int n_in,
                              void* d_out, int out_size, void* d_ws,
                              size_t ws_size, hipStream_t stream) {
  (void)in_sizes; (void)n_in; (void)out_size; (void)ws_size;
  const float* x      = (const float*)d_in[0];
  const float* w_qkv  = (const float*)d_in[1];
  const float* b_qkv  = (const float*)d_in[2];
  const float* w_proj = (const float*)d_in[3];
  const float* b_proj = (const float*)d_in[4];
  const float* w_fc1  = (const float*)d_in[5];
  const float* b_fc1  = (const float*)d_in[6];
  const float* w_fc2  = (const float*)d_in[7];
  const float* b_fc2  = (const float*)d_in[8];
  const float* g1     = (const float*)d_in[9];
  const float* bt1    = (const float*)d_in[10];
  const float* g2     = (const float*)d_in[11];
  const float* bt2    = (const float*)d_in[12];
  float* out = (float*)d_out;

  char* ws = (char*)d_ws;
  size_t off = 0;
  auto alloc = [&](size_t bytes) {
    void* p = ws + off;
    off += (bytes + 255) & ~(size_t)255;
    return p;
  };
  bf16_t* wt_qkv  = (bf16_t*)alloc((size_t)CDIM * 3 * CDIM * 2);
  bf16_t* wt_proj = (bf16_t*)alloc((size_t)CDIM * CDIM * 2);
  bf16_t* wt_fc1  = (bf16_t*)alloc((size_t)CDIM * 4 * CDIM * 2);
  bf16_t* wt_fc2  = (bf16_t*)alloc((size_t)4 * CDIM * CDIM * 2);
  bf16_t* ln1b    = (bf16_t*)alloc((size_t)MROWS * CDIM * 2);
  bf16_t* Qb      = (bf16_t*)alloc((size_t)MROWS * CDIM * 2);
  bf16_t* Kb      = (bf16_t*)alloc((size_t)MROWS * CDIM * 2);
  bf16_t* Vtb     = (bf16_t*)alloc((size_t)MROWS * CDIM * 2);
  bf16_t* Yb      = (bf16_t*)alloc((size_t)MROWS * CDIM * 2);
  float*  x1      = (float*) alloc((size_t)MROWS * CDIM * 4);
  bf16_t* ln2b    = (bf16_t*)alloc((size_t)MROWS * CDIM * 2);
  bf16_t* Hb      = (bf16_t*)alloc((size_t)MROWS * 4 * CDIM * 2);

  const dim3 tb32(32, 8);
  // weight convert/transpose: W[K][N] -> Wt[N][K] bf16
  wt_kernel<<<dim3(3 * CDIM / 32, CDIM / 32), tb32, 0, stream>>>(w_qkv, wt_qkv, CDIM, 3 * CDIM);
  wt_kernel<<<dim3(CDIM / 32, CDIM / 32), tb32, 0, stream>>>(w_proj, wt_proj, CDIM, CDIM);
  wt_kernel<<<dim3(4 * CDIM / 32, CDIM / 32), tb32, 0, stream>>>(w_fc1, wt_fc1, CDIM, 4 * CDIM);
  wt_kernel<<<dim3(CDIM / 32, 4 * CDIM / 32), tb32, 0, stream>>>(w_fc2, wt_fc2, 4 * CDIM, CDIM);

  // LN1
  ln_kernel<<<MROWS, 256, 0, stream>>>(x, g1, bt1, ln1b);

  // QKV GEMM with head-layout scatter epilogue
  gemm_bf16_kernel<3><<<dim3(3 * CDIM / BN, MROWS / BM), 256, 0, stream>>>(
      ln1b, wt_qkv, b_qkv, 3 * CDIM, CDIM, nullptr, nullptr, nullptr, Qb, Kb, Vtb);

  // causal attention
  attn_kernel<<<dim3(TSEQ / 64, NBATCH * NH), 128, 0, stream>>>(Qb, Kb, Vtb, Yb);

  // proj GEMM + residual (fp32 x1 = x + Y @ Wp + bp)
  gemm_bf16_kernel<2><<<dim3(CDIM / BN, MROWS / BM), 256, 0, stream>>>(
      Yb, wt_proj, b_proj, CDIM, CDIM, nullptr, x1, x, nullptr, nullptr, nullptr);

  // LN2
  ln_kernel<<<MROWS, 256, 0, stream>>>(x1, g2, bt2, ln2b);

  // FC1 + ReLU (bf16)
  gemm_bf16_kernel<1><<<dim3(4 * CDIM / BN, MROWS / BM), 256, 0, stream>>>(
      ln2b, wt_fc1, b_fc1, 4 * CDIM, CDIM, Hb, nullptr, nullptr, nullptr, nullptr, nullptr);

  // FC2 + residual -> fp32 output
  gemm_bf16_kernel<2><<<dim3(CDIM / BN, MROWS / BM), 256, 0, stream>>>(
      Hb, wt_fc2, b_fc2, CDIM, 4 * CDIM, nullptr, out, x1, nullptr, nullptr, nullptr);
}
